// PointLSTM_5411658793012
// MI455X (gfx1250) — matrix operand order repaired and checked by reading, not compiled, not run on hardware
//
#include <hip/hip_runtime.h>
#include <math.h>

typedef __bf16 bf16_t;
typedef __attribute__((ext_vector_type(16))) __bf16 v16bf;
typedef __attribute__((ext_vector_type(8)))  __bf16 v8bf;
typedef __attribute__((ext_vector_type(8)))  float  v8f;

#define NB_  4      // batch
#define NT   32     // time
#define NC   132    // input channels
#define NP   512    // points
#define NK   16     // neighbors
#define NH   256    // hidden
#define OCH  1024   // 4*NH gate rows
#define CIN  392    // 132 + 4 + 256
#define KPAD_NB 288 // 4 + 256 padded to 32
#define KPAD_BS 160 // 132 padded to 32
#define KB_NB (KPAD_NB/32)
#define KB_BS (KPAD_BS/32)
#define LDS_PAD 8   // bf16 pad per LDS column -> 148-dword stride, conflict-free

// ---------------------------------------------------------------------------
// Pre-pack the weight matrix into wave32 WMMA A-fragment layout (bf16).
// A-fragment (16x32 bf16) per lane: lane L holds M = L&15, K-half = L>>4;
// vgpr v<4  : K = kb*32 + half*8 + 2v + {0,1}
// vgpr v>=4 : K = kb*32 + 16 + half*8 + 2(v-4) + {0,1}
// base pack: K<132 -> Wc[o,k] - (k<4 ? Woff[o,k] : 0); else 0
// nb   pack: K<4 -> Woff[o,k]; K<260 -> Wh[o,k-4]; else 0
// ---------------------------------------------------------------------------
__global__ void prep_weights(const float* __restrict__ w,
                             bf16_t* __restrict__ Abs, bf16_t* __restrict__ Anb) {
    const int total_bs = 64 * KB_BS * 32 * 16;
    const int total_nb = 64 * KB_NB * 32 * 16;
    for (int i = blockIdx.x * blockDim.x + threadIdx.x; i < total_bs + total_nb;
         i += gridDim.x * blockDim.x) {
        bool is_nb = (i >= total_bs);
        int idx = is_nb ? (i - total_bs) : i;
        int KBn = is_nb ? KB_NB : KB_BS;
        int e    = idx & 15;
        int lane = (idx >> 4) & 31;
        int kb   = (idx >> 9) % KBn;
        int mt   = (idx >> 9) / KBn;
        int m = lane & 15, half = lane >> 4;
        int v = e >> 1, hl = e & 1;
        int k = kb * 32 + ((v < 4) ? (half * 8 + v * 2 + hl)
                                   : (16 + half * 8 + (v - 4) * 2 + hl));
        int o = mt * 16 + m;
        float val = 0.f;
        if (is_nb) {
            if (k < 4)        val = w[o * CIN + 132 + k];
            else if (k < 260) val = w[o * CIN + 136 + (k - 4)];
        } else {
            if (k < 132) {
                val = w[o * CIN + k];
                if (k < 4) val -= w[o * CIN + 132 + k];
            }
        }
        (is_nb ? Anb : Abs)[idx] = (bf16_t)val;
    }
}

// ---------------------------------------------------------------------------
// 16-NN per (b,t): query frame t vs ref frame (t-1, or t for t==0), 3D coords.
// Ref points staged in LDS; per-thread sorted top-16 in registers.
// ---------------------------------------------------------------------------
__global__ void __launch_bounds__(NP) knn_kernel(const float* __restrict__ in,
                                                 int* __restrict__ gi) {
    __shared__ float rx[NP], ry[NP], rz[NP];
    int bt = blockIdx.x;
    int b = bt >> 5, t = bt & 31;
    int tr = t ? (t - 1) : 0;
    const float* q = in + (((size_t)b * NT + t)  * NC) * NP;
    const float* r = in + (((size_t)b * NT + tr) * NC) * NP;
    int tid = threadIdx.x;
    rx[tid] = r[tid]; ry[tid] = r[NP + tid]; rz[tid] = r[2 * NP + tid];
    __syncthreads();
    float qx = q[tid], qy = q[NP + tid], qz = q[2 * NP + tid];
    float dist[NK]; int idx[NK];
#pragma unroll
    for (int j = 0; j < NK; j++) { dist[j] = 3.0e38f; idx[j] = 0; }
    for (int m = 0; m < NP; m++) {
        float dx = qx - rx[m], dy = qy - ry[m], dz = qz - rz[m];
        float d = dx * dx + dy * dy + dz * dz;
        if (d < dist[NK - 1]) {
            dist[NK - 1] = d; idx[NK - 1] = m;
#pragma unroll
            for (int j = NK - 1; j > 0; j--) {
                if (dist[j] < dist[j - 1]) {
                    float td = dist[j]; dist[j] = dist[j - 1]; dist[j - 1] = td;
                    int   ti = idx[j];  idx[j]  = idx[j - 1];  idx[j - 1]  = ti;
                }
            }
        }
    }
    int* dst = gi + (((size_t)bt) * NP + tid) * NK;
#pragma unroll
    for (int j = 0; j < NK; j++) dst[j] = idx[j];
}

// zero h0/c0 and copy position channels into layer_output[..,0:4,..]
__global__ void init_kernel(const float* __restrict__ in, float* __restrict__ h0,
                            float* __restrict__ c0, float* __restrict__ lo) {
    int i0 = blockIdx.x * blockDim.x + threadIdx.x;
    int stride = gridDim.x * blockDim.x;
    const int nh = NB_ * NP * NH;
    for (int j = i0; j < nh; j += stride) { h0[j] = 0.f; c0[j] = 0.f; }
    const int np = NB_ * NT * 4 * NP;
    for (int j = i0; j < np; j += stride) {
        int n = j & (NP - 1);
        int c = (j >> 9) & 3;
        int bt = j >> 11;
        lo[((size_t)bt * 260 + c) * NP + n] = in[((size_t)bt * NC + c) * NP + n];
    }
}

// Build bf16 K-contiguous column buffers for both GEMMs (per point m).
__global__ void stage_kernel(const float* __restrict__ in, const float* __restrict__ hprev,
                             bf16_t* __restrict__ Bnb, bf16_t* __restrict__ Bbs, int t) {
    int m = blockIdx.x, b = blockIdx.y;
    int tp = t ? (t - 1) : 0;
    const float* x  = in + (((size_t)b * NT + t)  * NC) * NP;
    const float* pp = in + (((size_t)b * NT + tp) * NC) * NP;
    const float* h  = hprev + ((size_t)b * NP + m) * NH;
    bf16_t* dn = Bnb + ((size_t)b * NP + m) * KPAD_NB;
    bf16_t* db = Bbs + ((size_t)b * NP + m) * KPAD_BS;
    for (int c = threadIdx.x; c < KPAD_NB; c += blockDim.x) {
        float v = 0.f;
        if (c < 4)        v = pp[c * NP + m];
        else if (c < 260) v = h[c - 4];
        dn[c] = (bf16_t)v;
    }
    for (int c = threadIdx.x; c < KPAD_BS; c += blockDim.x) {
        float v = (c < NC) ? x[c * NP + m] : 0.f;
        db[c] = (bf16_t)v;
    }
}

// ---------------------------------------------------------------------------
// GEMM: out[b][n][o] (transposed, f32) = A(1024 x Kpad, prepacked bf16) * B.
// One wave per 16x16 output tile, 8 waves per block.
// The shared 16-column B tile (contiguous in global) is staged ONCE per block
// into LDS via global_load_async_to_lds_b128 (ASYNCcnt), with +8 bf16 column
// padding (148-dword stride -> bank-conflict-free ds_loads). A streams from
// global (L2-resident prepack). v_wmma_f32_16x16x32_bf16 accumulates in f32.
// ---------------------------------------------------------------------------
__global__ void __launch_bounds__(256)
gemm_bf16(const bf16_t* __restrict__ Ap, const bf16_t* __restrict__ Bs,
          float* __restrict__ out, int KBcnt, int Kpad) {
    __shared__ __attribute__((aligned(32))) bf16_t Btile[16 * (KPAD_NB + LDS_PAD)];
    const int tid  = threadIdx.x;
    const int lane = tid & 31, wave = tid >> 5;
    const int nt = blockIdx.y, b = blockIdx.z;
    const int ldsK = Kpad + LDS_PAD;          // bf16 column stride in LDS
    const int cpc  = Kpad >> 3;               // 16B chunks per column

    // --- async-stage B tile: 16 columns x Kpad bf16 -> LDS (padded columns)
    const bf16_t* tile_base = Bs + ((size_t)b * NP + nt * 16) * Kpad;
    for (int c = tid; c < 16 * cpc; c += 256) {
        int col = c / cpc, i = c - col * cpc;
        // low 32 bits of a generic pointer into LDS == workgroup LDS byte offset
        unsigned lds_off = (unsigned)(size_t)(&Btile[col * ldsK + i * 8]);
        unsigned long long gaddr = (unsigned long long)(size_t)(tile_base + col * Kpad + i * 8);
        asm volatile("global_load_async_to_lds_b128 %0, %1, off"
                     :: "v"(lds_off), "v"(gaddr)
                     : "memory");
    }
    asm volatile("s_wait_asynccnt 0x0" ::: "memory");
    __syncthreads();

    // --- WMMA main loop: A from global, B fragments from LDS
    const int mt = blockIdx.x * 8 + wave;
    const bf16_t* ap = Ap + ((size_t)mt * KBcnt * 32 + lane) * 16;
    const bf16_t* bp = &Btile[(lane & 15) * ldsK + (lane >> 4) * 8];
    v8f acc = {0.f, 0.f, 0.f, 0.f, 0.f, 0.f, 0.f, 0.f};
    for (int kb = 0; kb < KBcnt; ++kb) {
        v16bf a  = *(const v16bf*)ap;
        v8bf  b0 = *(const v8bf*)bp;
        v8bf  b1 = *(const v8bf*)(bp + 16);
        v16bf bb = __builtin_shufflevector(b0, b1, 0, 1, 2, 3, 4, 5, 6, 7,
                                           8, 9, 10, 11, 12, 13, 14, 15);
        acc = __builtin_amdgcn_wmma_f32_16x16x32_bf16(
            false, a, false, bb, (short)0, acc, false, false);
        ap += 32 * 16;
        bp += 32;
    }
    // D layout: lanes 0-15 vgpr r -> (M=r, N=lane); lanes 16-31 -> (M=8+r, N=lane-16)
    int n  = nt * 16 + (lane & 15);
    int o0 = mt * 16 + (lane >> 4) * 8;
    float* dst = out + ((size_t)b * NP + n) * OCH + o0;
    *(float4*)(dst)     = make_float4(acc[0], acc[1], acc[2], acc[3]);
    *(float4*)(dst + 4) = make_float4(acc[4], acc[5], acc[6], acc[7]);
}

// ---------------------------------------------------------------------------
// LSTM gather + gates + max over neighbors. Block = 256 hid channels (lanes
// contiguous over o => coalesced gathers of nb/c rows).
// ---------------------------------------------------------------------------
__global__ void __launch_bounds__(NH)
lstm_kernel(const float* __restrict__ baseo, const float* __restrict__ nbo,
            const float* __restrict__ cprev, const int* __restrict__ gi,
            const float* __restrict__ bias,
            float* __restrict__ hnext, float* __restrict__ cnext,
            float* __restrict__ lo, float* __restrict__ hf, float* __restrict__ cf,
            int t) {
    int n = blockIdx.x, b = blockIdx.y, o = threadIdx.x;
    const float* bs = baseo + ((size_t)b * NP + n) * OCH;
    float bi  = bs[o]          + bias[o];
    float bff = bs[NH + o]     + bias[NH + o];
    float bo_ = bs[2 * NH + o] + bias[2 * NH + o];
    float bg  = bs[3 * NH + o] + bias[3 * NH + o];
    const int* id = gi + ((((size_t)b * NT + t) * NP) + n) * NK;
    float hmax = -3.0e38f, cmax = -3.0e38f;
    for (int j = 0; j < NK; j++) {
        int m = id[j];
        const float* nb = nbo + ((size_t)b * NP + m) * OCH;
        float g_i = 1.f / (1.f + __expf(-(bi  + nb[o])));
        float g_f = 1.f / (1.f + __expf(-(bff + nb[NH + o])));
        float g_o = 1.f / (1.f + __expf(-(bo_ + nb[2 * NH + o])));
        float g_g = tanhf(bg + nb[3 * NH + o]);
        float cg  = cprev[((size_t)b * NP + m) * NH + o];
        float cn = g_f * cg + g_i * g_g;
        float hn = g_o * tanhf(cn);
        hmax = fmaxf(hmax, hn);
        cmax = fmaxf(cmax, cn);
    }
    hnext[((size_t)b * NP + n) * NH + o] = hmax;
    cnext[((size_t)b * NP + n) * NH + o] = cmax;
    lo[(((size_t)b * NT + t) * 260 + 4 + o) * NP + n] = hmax;
    if (t == NT - 1) {
        hf[((size_t)b * NH + o) * NP + n] = hmax;
        cf[((size_t)b * NH + o) * NP + n] = cmax;
    }
}

extern "C" void kernel_launch(void* const* d_in, const int* in_sizes, int n_in,
                              void* d_out, int out_size, void* d_ws, size_t ws_size,
                              hipStream_t stream) {
    (void)in_sizes; (void)n_in; (void)out_size; (void)ws_size;
    const float* in   = (const float*)d_in[0];
    const float* w    = (const float*)d_in[1];
    const float* bias = (const float*)d_in[2];
    float* out = (float*)d_out;
    char*  ws  = (char*)d_ws;

    size_t off = 0;
    bf16_t* Abs = (bf16_t*)(ws + off); off += (size_t)64 * KB_BS * 32 * 16 * 2;
    bf16_t* Anb = (bf16_t*)(ws + off); off += (size_t)64 * KB_NB * 32 * 16 * 2;
    bf16_t* Bbs = (bf16_t*)(ws + off); off += (size_t)NB_ * NP * KPAD_BS * 2;
    bf16_t* Bnb = (bf16_t*)(ws + off); off += (size_t)NB_ * NP * KPAD_NB * 2;
    float* baseo = (float*)(ws + off); off += (size_t)NB_ * NP * OCH * 4;
    float* nbo   = (float*)(ws + off); off += (size_t)NB_ * NP * OCH * 4;
    float* hb[2], *cb[2];
    hb[0] = (float*)(ws + off); off += (size_t)NB_ * NP * NH * 4;
    hb[1] = (float*)(ws + off); off += (size_t)NB_ * NP * NH * 4;
    cb[0] = (float*)(ws + off); off += (size_t)NB_ * NP * NH * 4;
    cb[1] = (float*)(ws + off); off += (size_t)NB_ * NP * NH * 4;

    // Output layout: layer_output | h_f | c_f | group_ind
    const size_t LO_SZ = (size_t)NB_ * NT * 260 * NP;
    float* lo = out;
    float* hf = out + LO_SZ;
    float* cf = hf + (size_t)NB_ * NH * NP;
    int*   gi = (int*)(cf + (size_t)NB_ * NH * NP);

    prep_weights<<<256, 256, 0, stream>>>(w, Abs, Anb);
    knn_kernel<<<NB_ * NT, NP, 0, stream>>>(in, gi);
    init_kernel<<<512, 256, 0, stream>>>(in, hb[0], cb[0], lo);

    for (int t = 0; t < NT; ++t) {
        int cur = t & 1, nxt = cur ^ 1;
        stage_kernel<<<dim3(NP, NB_), 128, 0, stream>>>(in, hb[cur], Bnb, Bbs, t);
        gemm_bf16<<<dim3(8, 32, NB_), 256, 0, stream>>>(Anb, Bnb, nbo, KB_NB, KPAD_NB);
        gemm_bf16<<<dim3(8, 32, NB_), 256, 0, stream>>>(Abs, Bbs, baseo, KB_BS, KPAD_BS);
        lstm_kernel<<<dim3(NP, NB_), NH, 0, stream>>>(baseo, nbo, cb[cur], gi, bias,
                                                      hb[nxt], cb[nxt], lo, hf, cf, t);
    }
}
